// CausalWeighting_72756745994823
// MI455X (gfx1250) — matrix operand order repaired and checked by reading, not compiled, not run on hardware
//
#include <hip/hip_runtime.h>

#define BATCH 8
#define NNODE 512
#define DFEAT 256
#define NEDGE 16384
#define KDIM  512     // 2*D
#define HDIM  256     // hidden width
#define ROWS  32      // (batch,edge) rows per block
#define BLOCK 256     // 8 waves of 32

typedef __attribute__((ext_vector_type(16))) __bf16 v16bf;
typedef __attribute__((ext_vector_type(8)))  float  v8f;

union BF16Frag {
    v16bf          v;
    uint4          q[2];
    unsigned short u[16];
};

__device__ __forceinline__ unsigned short f32_to_bf16(float f) {
    unsigned int x = __float_as_uint(f);
    unsigned int r = 0x7FFFu + ((x >> 16) & 1u);   // round to nearest even
    return (unsigned short)((x + r) >> 16);
}

// ---------------------------------------------------------------------------
// Zero the (B, N, N) output (harness poisons it with 0xAA).
// 2,097,152 floats -> 524,288 float4 stores.
__global__ __launch_bounds__(256) void zero_out_kernel(float4* __restrict__ out) {
    out[(size_t)blockIdx.x * 256 + threadIdx.x] = make_float4(0.f, 0.f, 0.f, 0.f);
}

// ---------------------------------------------------------------------------
// Pack W1 (f32 [512][256]) into bf16 WMMA-B fragment layout in workspace.
// Fragment f = kt*16 + nt covers K in [kt*32, kt*32+32), Ncol in [nt*16, nt*16+16).
// Within a fragment: lane L holds Ncol = nt*16 + (L%16),
//                    K = kt*32 + (L/16)*16 + idx, idx = 0..15 (16 bf16 = 32 bytes).
__global__ __launch_bounds__(256) void pack_w1_kernel(const float* __restrict__ W1,
                                                      unsigned short* __restrict__ Bws) {
    const int t      = blockIdx.x * 256 + threadIdx.x;  // 0 .. 131071
    const int f      = t >> 9;                           // fragment 0..255
    const int within = t & 511;
    const int lane   = within >> 4;
    const int idx    = within & 15;
    const int kt     = f >> 4;
    const int nt     = f & 15;
    const int K      = kt * 32 + ((lane >> 4) << 4) + idx;
    const int Ncol   = nt * 16 + (lane & 15);
    Bws[t] = f32_to_bf16(W1[K * HDIM + Ncol]);
}

// ---------------------------------------------------------------------------
// Main fused kernel: gather pair rows -> bf16 LDS, WMMA GEMM (K=512, Nout=256),
// fused bias+ReLU+W2 dot + sigmoid epilogue, scatter to out[b, src, dst].
__global__ __launch_bounds__(BLOCK) void
gemm_edge_kernel(const float* __restrict__ feat,
                 const float* __restrict__ b1,
                 const float* __restrict__ W2,
                 const float* __restrict__ b2,
                 const long long* __restrict__ eidx,
                 const unsigned short* __restrict__ Bws,
                 float* __restrict__ out)
{
    // Padded by 8 bf16 (16B) per row: row stride 1040B -> conflict-free b128 reads.
    __shared__ __align__(16) unsigned short Abuf[ROWS][KDIM + 8];
    __shared__ float rowsum[ROWS];

    const int tid  = threadIdx.x;
    const int row0 = blockIdx.x * ROWS;

    if (tid < ROWS) rowsum[tid] = 0.f;

    // ---- Stage A: 32 rows x 512 bf16. 8 threads per row, 64 floats each.
    {
        const int row  = tid >> 3;      // 0..31
        const int seg  = tid & 7;       // 0..7 (segs 0-3 = fi, 4-7 = fj)
        const int R    = row0 + row;
        const int b    = R >> 14;                // E = 2^14
        const int e    = R & (NEDGE - 1);
        const int node = (seg < 4) ? (int)eidx[e] : (int)eidx[NEDGE + e];
        const float4* src4 =
            (const float4*)(feat + ((size_t)b * NNODE + node) * DFEAT + (seg & 3) * 64);
        const int kbase = seg * 64;
#pragma unroll
        for (int j = 0; j < 16; ++j) {
            float4 v = src4[j];
            ushort4 h;
            h.x = f32_to_bf16(v.x);
            h.y = f32_to_bf16(v.y);
            h.z = f32_to_bf16(v.z);
            h.w = f32_to_bf16(v.w);
            *(ushort4*)&Abuf[row][kbase + j * 4] = h;
        }
    }
    __syncthreads();

    const int wave = tid >> 5;
    const int lane = tid & 31;
    const int mt   = wave >> 2;   // row-tile 0..1 (16 rows each)
    const int cg   = wave & 3;    // column group -> tiles cg*4 .. cg*4+3

    v8f acc[4] = {};

    // A-fragment addressing (16-bit A 16x32 layout):
    // lane L -> M = L%16, Kbase = (L/16)*16; regs 0-7: K=Kbase+0..15
    const int arow  = mt * 16 + (lane & 15);
    const int akoff = (lane >> 4) << 4;

    for (int kt = 0; kt < 16; ++kt) {
        BF16Frag a;
        const unsigned short* ap = &Abuf[arow][kt * 32 + akoff];
        a.q[0] = *(const uint4*)(ap);
        a.q[1] = *(const uint4*)(ap + 8);

        const unsigned short* bp = Bws + ((size_t)(kt * 16 + cg * 4) << 9) + lane * 16;
#pragma unroll
        for (int t = 0; t < 4; ++t) {
            BF16Frag bb;
            const unsigned short* bt = bp + (t << 9);
            bb.q[0] = *(const uint4*)(bt);
            bb.q[1] = *(const uint4*)(bt + 8);
            acc[t] = __builtin_amdgcn_wmma_f32_16x16x32_bf16(
                false, a.v, false, bb.v, (short)0, acc[t], false, false);
        }
    }

    // ---- Fused epilogue: p[r] = sum_over_cols relu(h + b1) * W2
    // C/D layout: acc[t][r] at lane L = h[M][Ncol], M = r + 8*(L/16), Ncol = tile*16 + L%16
    float p[8];
#pragma unroll
    for (int r = 0; r < 8; ++r) p[r] = 0.f;
#pragma unroll
    for (int t = 0; t < 4; ++t) {
        const int   ncol = (cg * 4 + t) * 16 + (lane & 15);
        const float bias = b1[ncol];
        const float wv   = W2[ncol];
#pragma unroll
        for (int r = 0; r < 8; ++r) {
            float h = acc[t][r] + bias;
            h = h > 0.f ? h : 0.f;
            p[r] += h * wv;
        }
    }

    // Reduce each p[r] across the 16 lanes of its half-wave, then across waves via LDS.
#pragma unroll
    for (int r = 0; r < 8; ++r) {
        float v = p[r];
        v += __shfl_xor(v, 1, 16);
        v += __shfl_xor(v, 2, 16);
        v += __shfl_xor(v, 4, 16);
        v += __shfl_xor(v, 8, 16);
        if ((lane & 15) == 0)
            atomicAdd(&rowsum[mt * 16 + ((lane >> 4) << 3) + r], v);
    }
    __syncthreads();

    // ---- Sigmoid + scatter: one thread per row.
    if (tid < ROWS) {
        const int R = row0 + tid;
        const int b = R >> 14;
        const int e = R & (NEDGE - 1);
        const int s = (int)eidx[e];
        const int d = (int)eidx[NEDGE + e];
        const float x   = rowsum[tid] + b2[0];
        const float sig = 1.f / (1.f + __expf(-x));
        out[((size_t)b << 18) + ((size_t)s << 9) + (size_t)d] = sig;
    }
}

// ---------------------------------------------------------------------------
extern "C" void kernel_launch(void* const* d_in, const int* in_sizes, int n_in,
                              void* d_out, int out_size, void* d_ws, size_t ws_size,
                              hipStream_t stream) {
    const float*     feat = (const float*)d_in[0];      // (8, 512, 256) f32
    const float*     W1   = (const float*)d_in[1];      // (512, 256) f32
    const float*     b1   = (const float*)d_in[2];      // (256,) f32
    const float*     W2   = (const float*)d_in[3];      // (256, 1) f32
    const float*     b2   = (const float*)d_in[4];      // (1,) f32
    const long long* eidx = (const long long*)d_in[5];  // (2, 16384) i64
    float*           out  = (float*)d_out;              // (8, 512, 512) f32
    unsigned short*  Bws  = (unsigned short*)d_ws;      // 256 KB bf16 W1 fragments

    // (8*512*512)/4 float4 = 524288 -> 2048 blocks
    zero_out_kernel<<<2048, 256, 0, stream>>>((float4*)out);
    // 512*256 bf16 elems -> 512 blocks
    pack_w1_kernel<<<512, 256, 0, stream>>>(W1, Bws);
    // 131072 rows / 32 per block
    gemm_edge_kernel<<<(BATCH * NEDGE) / ROWS, BLOCK, 0, stream>>>(
        feat, b1, W2, b2, eidx, Bws, out);
}
